// CtrlPointHungarianMatcher_48438641164463
// MI455X (gfx1250) — matrix operand order repaired and checked by reading, not compiled, not run on hardware
//
#include <hip/hip_runtime.h>
#include <hip/hip_bf16.h>

typedef __attribute__((ext_vector_type(16))) _Float16 v16h;
typedef __attribute__((ext_vector_type(8)))  float    v8f;

#define N_TOT   2400      // bs*Q
#define T_TOT   160       // targets
#define P_TOT   12544     // sample points
#define MDIM    128       // mask H = W
#define MASKSZ  16384     // 128*128
#define KPD     32        // 2*n_ctrl
#define NCHUNK  392       // P / 32
#define TTILES  10        // T / 16
#define NBLOCKS 150       // N / 16

// ---- bilinear sample, align_corners=False, zero padding (matches reference) ----
__device__ __forceinline__ float bsample(const float* __restrict__ m, float cx, float cy) {
    float px = cx * (float)MDIM - 0.5f;
    float py = cy * (float)MDIM - 0.5f;
    float fx = floorf(px), fy = floorf(py);
    int x0 = (int)fx, y0 = (int)fy;
    float wx1 = px - fx, wy1 = py - fy;
    float wx0 = 1.0f - wx1, wy0 = 1.0f - wy1;
    int x1 = x0 + 1, y1 = y0 + 1;
    int x0c = min(max(x0, 0), MDIM - 1), y0c = min(max(y0, 0), MDIM - 1);
    int x1c = min(max(x1, 0), MDIM - 1), y1c = min(max(y1, 0), MDIM - 1);
    float vx0 = (x0 >= 0 && x0 < MDIM) ? 1.0f : 0.0f;
    float vy0 = (y0 >= 0 && y0 < MDIM) ? 1.0f : 0.0f;
    float vx1 = (x1 >= 0 && x1 < MDIM) ? 1.0f : 0.0f;
    float vy1 = (y1 >= 0 && y1 < MDIM) ? 1.0f : 0.0f;
    float v00 = m[y0c * MDIM + x0c] * (vx0 * vy0);
    float v10 = m[y0c * MDIM + x1c] * (vx1 * vy0);
    float v01 = m[y1c * MDIM + x0c] * (vx0 * vy1);
    float v11 = m[y1c * MDIM + x1c] * (vx1 * vy1);
    return v00 * (wx0 * wy0) + v10 * (wx1 * wy0) + v01 * (wx0 * wy1) + v11 * (wx1 * wy1);
}

__device__ __forceinline__ float sigmoidf_(float x) { return 1.0f / (1.0f + expf(-x)); }

// ---- phase 0: zero t-row-sum accumulator ----
__global__ void zero_tsum_kernel(float* p) {
    int i = blockIdx.x * blockDim.x + threadIdx.x;
    if (i < T_TOT) p[i] = 0.0f;
}

// ---- phase 1: sample tgt_masks at point_coords, emit B fragments in WMMA lane order ----
// Fragment layout per (chunk c, t-tile j): 32 lanes x 16 halves (32B).
// Lane L: column n = j*16 + (L&15); K offsets = c*32 + ((L&16)?16:0) + i, i=0..15.
__global__ __launch_bounds__(256) void tsamp_kernel(
    const float* __restrict__ tgt_masks, const float* __restrict__ coords,
    _Float16* __restrict__ tfrag, float* __restrict__ tsum)
{
    int wg = blockIdx.x * 8 + (threadIdx.x >> 5);       // 0..3919 = NCHUNK*TTILES
    if (wg >= NCHUNK * TTILES) return;
    int lane = threadIdx.x & 31;
    int c = wg / TTILES, j = wg - c * TTILES;
    int t = j * 16 + (lane & 15);
    int kbase = c * 32 + ((lane & 16) ? 16 : 0);
    const float* mk = tgt_masks + (size_t)t * MASKSZ;
    const float2* c2 = (const float2*)coords;
    union { v16h v; _Float16 h[16]; } f;
    float s = 0.0f;
#pragma unroll
    for (int i = 0; i < 16; ++i) {
        float2 cc = c2[kbase + i];
        float val = bsample(mk, cc.x, cc.y);
        s += val;
        f.h[i] = (_Float16)val;
    }
    *(v16h*)(tfrag + ((size_t)wg * 32 + lane) * 16) = f.v;
    atomicAdd(&tsum[t], s);
}

// ---- phase 2: fused sample + sigmoid + WMMA GEMM + full cost epilogue ----
// Block: 320 threads = 10 waves; covers 16 queries x 160 targets. Wave w = t-tile w.
// Double-buffered A tile -> one barrier per K-chunk; row sums carried in registers.
__global__ __launch_bounds__(320) void matcher_kernel(
    const float* __restrict__ logits, const float* __restrict__ ctrl,
    const float* __restrict__ segmask, const float* __restrict__ tgt_pts,
    const float* __restrict__ coords, const _Float16* __restrict__ tfrag,
    const float* __restrict__ tsum, const int* __restrict__ lvlp,
    float* __restrict__ out)
{
    __shared__ _Float16 ldsA[2][16 * 34];  // double-buffered 16x32 A tile, rows padded (17 dwords)
    __shared__ float ldsPts[16 * KPD];
    __shared__ float ldsLogit[16];
    __shared__ float ldsSSum[16];

    const int tid = threadIdx.x;
    const int lane = tid & 31;
    const int wv = __builtin_amdgcn_readfirstlane(tid >> 5);   // wave-uniform t-tile id
    const int nbase = blockIdx.x * 16;
    const float2* c2 = (const float2*)coords;

    for (int i = tid; i < 16 * KPD; i += 320) ldsPts[i] = ctrl[(size_t)nbase * KPD + i];
    if (tid < 16) { ldsLogit[tid] = logits[nbase + tid]; ldsSSum[tid] = 0.0f; }
    __syncthreads();

    // wave-uniform (SGPR) mask base pointers -> saddr-form gathers
    const float* __restrict__ mk0 = segmask + (size_t)(nbase + wv) * MASKSZ;
    const float* __restrict__ mk1 = segmask + (size_t)(nbase + wv + 10) * MASKSZ; // wv<6 only
    // per-lane B fragment pointer; advances by TTILES*512 halves per chunk
    const _Float16* __restrict__ bptr = tfrag + ((size_t)wv * 32 + lane) * 16;

    float rs0 = 0.0f, rs1 = 0.0f;
    v8f acc = {};
    for (int c = 0; c < NCHUNK; ++c) {
        _Float16* Abuf = ldsA[c & 1];
        // --- build A tile: s = sigmoid(bilinear(mask_row, coord)) ---
        float2 cc = c2[c * 32 + lane];
        float s0 = sigmoidf_(bsample(mk0, cc.x, cc.y));
        Abuf[wv * 34 + lane] = (_Float16)s0;
        rs0 += s0;
        if (wv < 6) {
            float s1 = sigmoidf_(bsample(mk1, cc.x, cc.y));
            Abuf[(wv + 10) * 34 + lane] = (_Float16)s1;
            rs1 += s1;
        }
        __syncthreads();   // implies s_wait_dscnt 0: prior reads of this buffer are done

        // --- A fragment from LDS, contiguous k = 16*(lane/16) + s (same order as B) ---
        union { v16h v; unsigned int d[8]; } A;
        const unsigned int* Ar = (const unsigned int*)Abuf + (lane & 15) * 17;
        int pb = (lane & 16) ? 8 : 0;
#pragma unroll
        for (int q = 0; q < 4; ++q) { A.d[q] = Ar[pb + q]; A.d[4 + q] = Ar[pb + 4 + q]; }

        // --- B fragment: one coalesced 32B load per lane (pre-swizzled) ---
        v16h B = *(const v16h*)(bptr + (size_t)c * (TTILES * 512));

        acc = __builtin_amdgcn_wmma_f32_16x16x32_f16(false, A.v, false, B,
                                                     (short)0, acc, false, false);
        // no trailing barrier: next iteration writes the other A buffer
    }

    // flush register row-sums once
    atomicAdd(&ldsSSum[wv], rs0);
    if (wv < 6) atomicAdd(&ldsSSum[wv + 10], rs1);
    __syncthreads();

    // --- epilogue: dice + focal class cost + L1 keypoint cdist ---
    const int tcol = wv * 16 + (lane & 15);
    const float tS = tsum[tcol];
    const float mw = (lvlp[0] < 2) ? 5.0f : 0.0f;   // MASK_W gated by lvl
    float tp[KPD];
    const float* tptr = tgt_pts + (size_t)tcol * KPD;
#pragma unroll
    for (int i = 0; i < KPD; ++i) tp[i] = tptr[i];

    const int mrow0 = (lane & 16) ? 8 : 0;
#pragma unroll
    for (int v = 0; v < 8; ++v) {
        int m = mrow0 + v;
        // focal class cost (GAMMA=2 -> square)
        float p = sigmoidf_(ldsLogit[m]);
        float pos = 0.25f * (1.0f - p) * (1.0f - p) * (-logf(p + 1e-8f));
        float neg = 0.75f * p * p * (-logf(1.0f - p + 1e-8f));
        float cclass = pos - neg;
        // L1 cdist over 32 dims
        float kd = 0.0f;
        const float* op = &ldsPts[m * KPD];
#pragma unroll
        for (int i = 0; i < KPD; ++i) kd += fabsf(op[i] - tp[i]);
        // dice cost
        float dice = 1.0f - (2.0f * acc[v] + 1.0f) / (ldsSSum[m] + tS + 1.0f);
        out[(size_t)(nbase + m) * T_TOT + tcol] = 2.0f * cclass + 5.0f * kd + mw * dice;
    }
}

extern "C" void kernel_launch(void* const* d_in, const int* in_sizes, int n_in,
                              void* d_out, int out_size, void* d_ws, size_t ws_size,
                              hipStream_t stream) {
    const float* pred_logits = (const float*)d_in[0];   // [8,300,1]
    const float* pred_ctrl   = (const float*)d_in[1];   // [8,300,16,2]
    const float* pred_seg    = (const float*)d_in[2];   // [8,300,128,128]
    const float* tgt_pts     = (const float*)d_in[3];   // [160,16,2]
    const float* tgt_masks   = (const float*)d_in[4];   // [160,128,128]
    const float* coords      = (const float*)d_in[5];   // [12544,2]
    const int*   lvl         = (const int*)d_in[6];     // scalar
    float* out = (float*)d_out;                         // [8,300,160]

    // workspace: [0,640) t row sums; [1024, 1024+4014080) pre-swizzled B fragments (f16)
    float*     tsum  = (float*)d_ws;
    _Float16*  tfrag = (_Float16*)((char*)d_ws + 1024);

    zero_tsum_kernel<<<1, 256, 0, stream>>>(tsum);
    tsamp_kernel<<<(NCHUNK * TTILES + 7) / 8, 256, 0, stream>>>(tgt_masks, coords, tfrag, tsum);
    matcher_kernel<<<NBLOCKS, 320, 0, stream>>>(pred_logits, pred_ctrl, pred_seg, tgt_pts,
                                                coords, tfrag, tsum, lvl, out);
}
